// MultiHeadAttention_3934190043898
// MI455X (gfx1250) — matrix operand order repaired and checked by reading, not compiled, not run on hardware
//
#include <hip/hip_runtime.h>
#include <hip/hip_bf16.h>
#include <math.h>

// ---- problem constants (from reference) ----
#define NHEAD   16
#define DMODEL  1024
#define DKH     64
#define BSZ     8
#define SEQL    1024
#define MTOT    (BSZ * SEQL)      // 8192 rows for all projections

typedef __attribute__((ext_vector_type(16))) __bf16 v16bf;
typedef __attribute__((ext_vector_type(8)))  __bf16 v8bf;
typedef __attribute__((ext_vector_type(8)))  float  v8f;

__device__ __forceinline__ v8f wmma_bf16(v16bf a, v16bf b, v8f c) {
  // D = A(16x32 bf16) * B(32x16 bf16) + C(16x16 f32)
  return __builtin_amdgcn_wmma_f32_16x16x32_bf16(
      /*neg_a=*/false, a, /*neg_b=*/false, b,
      /*c_mod=*/(short)0, c, /*reuse_a=*/false, /*reuse_b=*/false);
}

// Assemble an A-fragment (16x32) from a row-major source:
// lane m = lane&15, half lh = lane>>4.
// elements [0..7]  = src_row[lh*16 + 0..7]
// elements [8..15] = src_row[lh*16 + 8 + 0..7]
__device__ __forceinline__ v16bf make_a_frag(const __bf16* row_ptr, int lh) {
  v8bf lo = *(const v8bf*)(row_ptr + lh * 16);
  v8bf hi = *(const v8bf*)(row_ptr + lh * 16 + 8);
  v16bf a;
  #pragma unroll
  for (int i = 0; i < 8; ++i) { a[i] = lo[i]; a[8 + i] = hi[i]; }
  return a;
}

// generic pointer to a __shared__ object: low 32 bits are the LDS byte offset
__device__ __forceinline__ unsigned lds_off(const void* p) {
  return (unsigned)(unsigned long long)p;
}

// ------------------------------------------------------------------
// elementwise f32 -> bf16
__global__ __launch_bounds__(256)
void cvt_bf16_kernel(const float* __restrict__ src, __bf16* __restrict__ dst, int n) {
  int i = blockIdx.x * 256 + threadIdx.x;
  if (i < n) dst[i] = (__bf16)src[i];
}

// W (K x N row-major f32) -> Wt (N x K row-major bf16)
__global__ __launch_bounds__(256)
void cvt_wt_kernel(const float* __restrict__ W, __bf16* __restrict__ Wt) {
  int i = blockIdx.x * 256 + threadIdx.x;     // 0 .. 1M-1
  int k = i >> 10, n = i & 1023;
  Wt[(size_t)n * DMODEL + k] = (__bf16)W[(size_t)k * DMODEL + n];
}

// ------------------------------------------------------------------
// GEMM: C(M x 1024) = A(M x 1024 bf16) * Wt^T + bias
//   Wt is N x K row-major (i.e. W transposed), K = N = 1024.
// Double-buffered LDS pipeline fed by GLOBAL_LOAD_ASYNC_TO_LDS_B128
// (ASYNCcnt-tracked direct memory->LDS copies, no VGPR staging).
// MODE 0: write bf16 into [b, h, s, dk]      (Q, K projections)
// MODE 1: write bf16 into [b, h, dk, s]      (V projection, per-head transposed)
// MODE 2: write f32 row-major M x 1024       (output projection -> d_out)
template <int MODE>
__global__ __launch_bounds__(256)
void gemm_bf16_kernel(const __bf16* __restrict__ A,
                      const __bf16* __restrict__ Bt,
                      const float*  __restrict__ bias,
                      void* __restrict__ dstv) {
  // 48-elem row stride: 96B rows keep 16B alignment for b128 chunks
  __shared__ __align__(16) __bf16 As[2][128][48];
  __shared__ __align__(16) __bf16 Bs[2][128][48];

  const int tid  = threadIdx.x;
  const int lane = tid & 31;
  const int wave = tid >> 5;          // 0..7
  const int ln   = lane & 15;
  const int lh   = lane >> 4;         // lane half
  const int wr   = wave >> 1;         // 0..3 : 32-row strips
  const int wc   = wave & 1;          // 0..1 : 64-col strips
  const int m0   = blockIdx.x * 128;
  const int n0   = blockIdx.y * 128;

  const int ldr = tid >> 1;           // 0..127 tile row this thread stages
  const int ldh = (tid & 1) * 16;     // which 16-element half of the 32-wide K slab

  const __bf16* agp = A  + (size_t)(m0 + ldr) * 1024 + ldh;
  const __bf16* bgp = Bt + (size_t)(n0 + ldr) * 1024 + ldh;

  // issue this thread's 32B share of the A and Bt tiles for k-slab k0 into buf
  auto issue_stage = [&](int buf, int k0) {
    unsigned la = lds_off(&As[buf][ldr][ldh]);
    unsigned lb = lds_off(&Bs[buf][ldr][ldh]);
    const __bf16* ga = agp + k0;
    const __bf16* gb = bgp + k0;
    asm volatile(
        "global_load_async_to_lds_b128 %0, %1, off\n\t"
        "global_load_async_to_lds_b128 %0, %1, off offset:16\n\t"
        "global_load_async_to_lds_b128 %2, %3, off\n\t"
        "global_load_async_to_lds_b128 %2, %3, off offset:16"
        :: "v"(la), "v"(ga), "v"(lb), "v"(gb) : "memory");
  };

  v8f acc[2][4] = {};
  issue_stage(0, 0);

  for (int k0 = 0; k0 < 1024; k0 += 32) {
    const int buf = (k0 >> 5) & 1;
    // drain this wave's async copies for the current buffer
    asm volatile("s_wait_asynccnt 0x0" ::: "memory");
    __syncthreads();   // all waves' copies landed; prior buffer no longer read
    if (k0 + 32 < 1024) issue_stage(buf ^ 1, k0 + 32);  // overlap with compute

    v16bf af[2];
    #pragma unroll
    for (int r = 0; r < 2; ++r)
      af[r] = make_a_frag(&As[buf][wr * 32 + r * 16 + ln][0], lh);

    v16bf bfrag[4];
    #pragma unroll
    for (int c = 0; c < 4; ++c)   // B frag: lane col = ln, 16 contiguous K at lh*16
      bfrag[c] = *(const v16bf*)&Bs[buf][wc * 64 + c * 16 + ln][lh * 16];

    #pragma unroll
    for (int r = 0; r < 2; ++r)
      #pragma unroll
      for (int c = 0; c < 4; ++c)
        acc[r][c] = wmma_bf16(af[r], bfrag[c], acc[r][c]);
  }

  // epilogue: C layout => row = lh*8 + v, col = ln (per 16x16 tile)
  #pragma unroll
  for (int r = 0; r < 2; ++r) {
    const int gr0 = m0 + wr * 32 + r * 16 + lh * 8;
    #pragma unroll
    for (int c = 0; c < 4; ++c) {
      const int gc = n0 + wc * 64 + c * 16 + ln;
      const float bv = bias[gc];
      #pragma unroll
      for (int v = 0; v < 8; ++v) {
        const int gr = gr0 + v;
        const float val = acc[r][c][v] + bv;
        if constexpr (MODE == 2) {
          ((float*)dstv)[(size_t)gr * DMODEL + gc] = val;
        } else {
          const int b  = gr >> 10, s  = gr & 1023;
          const int hd = gc >> 6,  dk = gc & 63;
          const size_t bh = (size_t)b * NHEAD + hd;
          if constexpr (MODE == 0)
            ((__bf16*)dstv)[(bh * SEQL + s) * DKH + dk] = (__bf16)val;
          else
            ((__bf16*)dstv)[(bh * DKH + dk) * SEQL + s] = (__bf16)val;
        }
      }
    }
  }
}

// ------------------------------------------------------------------
// Flash-style attention. Each wave owns 16 query rows of one (b,h).
// Streams 32-key tiles: S = Q Kt (4 WMMA), online softmax, O += P V (4 WMMA).
__global__ __launch_bounds__(128)
void attention_kernel(const __bf16* __restrict__ Qh,   // [b,h,s,dk] bf16
                      const __bf16* __restrict__ Kh,   // [b,h,s,dk] bf16
                      const __bf16* __restrict__ Vt,   // [b,h,dk,s] bf16
                      const int*    __restrict__ mask, // [b,s,s]
                      __bf16* __restrict__ ctx) {      // [b,s,h*64+dk] bf16
  __shared__ __align__(32) __bf16 plds[4][16][32];     // per-wave P bounce buffer

  const int lane = threadIdx.x & 31;
  const int wave = threadIdx.x >> 5;      // 0..3
  const int ln   = lane & 15;
  const int lh   = lane >> 4;
  const int head  = blockIdx.y;
  const int batch = blockIdx.z;
  const int q0    = blockIdx.x * 64 + wave * 16;
  const size_t bh = (size_t)batch * NHEAD + head;

  const __bf16* Qbase = Qh + (bh * SEQL + q0) * DKH;
  const __bf16* Kbase = Kh + bh * SEQL * DKH;
  const __bf16* Vbase = Vt + bh * DKH * SEQL;
  const int*    mbase = mask + ((size_t)batch * SEQL + q0) * SEQL;

  // Q fragments for dk-chunks 0-31 and 32-63 (reused across all key tiles)
  v16bf qa[2];
  #pragma unroll
  for (int d = 0; d < 2; ++d)
    qa[d] = make_a_frag(Qbase + (size_t)ln * DKH + d * 32, lh);

  v8f o[4] = {};
  float mrun[8], lrun[8];
  #pragma unroll
  for (int v = 0; v < 8; ++v) { mrun[v] = -1e30f; lrun[v] = 0.f; }
  const float scale = 0.125f;   // 1/sqrt(64)

  for (int k0 = 0; k0 < SEQL; k0 += 32) {
    // prefetch next key tile (global_prefetch_b8) to cover L2 latency
    if (k0 + 32 < SEQL) {
      __builtin_prefetch(Kbase + (size_t)(k0 + 32 + ln) * DKH, 0, 3);
      __builtin_prefetch(Vbase + (size_t)ln * SEQL + k0 + 32, 0, 3);
    }

    // ---- S = Q K^T for this 16q x 32k tile ----
    v8f s0 = {}, s1 = {};
    #pragma unroll
    for (int d = 0; d < 2; ++d) {
      v16bf kb0 = *(const v16bf*)(Kbase + (size_t)(k0 + ln)      * DKH + d * 32 + lh * 16);
      v16bf kb1 = *(const v16bf*)(Kbase + (size_t)(k0 + 16 + ln) * DKH + d * 32 + lh * 16);
      s0 = wmma_bf16(qa[d], kb0, s0);
      s1 = wmma_bf16(qa[d], kb1, s1);
    }

    // ---- mask + scale + online softmax (rows = lh*8+v across 16 lanes) ----
    float p0[8], p1[8], alpha[8];
    #pragma unroll
    for (int v = 0; v < 8; ++v) {
      const int qrow = lh * 8 + v;
      const int moff = qrow * SEQL + k0;
      float t0 = (mbase[moff + ln]      != 0) ? s0[v] * scale : -1e9f;
      float t1 = (mbase[moff + 16 + ln] != 0) ? s1[v] * scale : -1e9f;
      float rm = fmaxf(t0, t1);
      rm = fmaxf(rm, __shfl_xor(rm, 1));
      rm = fmaxf(rm, __shfl_xor(rm, 2));
      rm = fmaxf(rm, __shfl_xor(rm, 4));
      rm = fmaxf(rm, __shfl_xor(rm, 8));
      const float mnew = fmaxf(mrun[v], rm);
      const float a    = __expf(mrun[v] - mnew);
      mrun[v] = mnew;
      alpha[v] = a;
      const float e0 = __expf(t0 - mnew);
      const float e1 = __expf(t1 - mnew);
      float rs = e0 + e1;
      rs += __shfl_xor(rs, 1);
      rs += __shfl_xor(rs, 2);
      rs += __shfl_xor(rs, 4);
      rs += __shfl_xor(rs, 8);
      lrun[v] = lrun[v] * a + rs;
      p0[v] = e0; p1[v] = e1;
    }

    // rescale running O
    #pragma unroll
    for (int t = 0; t < 4; ++t)
      #pragma unroll
      for (int v = 0; v < 8; ++v) o[t][v] *= alpha[v];

    // ---- redistribute P: C layout -> A layout via per-wave LDS ----
    #pragma unroll
    for (int v = 0; v < 8; ++v) {
      plds[wave][lh * 8 + v][ln]      = (__bf16)p0[v];
      plds[wave][lh * 8 + v][16 + ln] = (__bf16)p1[v];
    }
    __builtin_amdgcn_wave_barrier();
    asm volatile("s_wait_dscnt 0" ::: "memory");   // DS ops are wave-in-order; wait for stores
    v16bf pa = make_a_frag(&plds[wave][ln][0], lh);
    __builtin_amdgcn_wave_barrier();

    // ---- O += P * V  (4 dk col-tiles) ----
    #pragma unroll
    for (int t = 0; t < 4; ++t) {
      v16bf vb = *(const v16bf*)(Vbase + (size_t)(t * 16 + ln) * SEQL + k0 + lh * 16);
      o[t] = wmma_bf16(pa, vb, o[t]);
    }
  }

  // normalize and write ctx in [b, s, head*64+dk] (bf16, feeds output GEMM)
  float inv[8];
  #pragma unroll
  for (int v = 0; v < 8; ++v) inv[v] = 1.0f / lrun[v];
  __bf16* cb = ctx + ((size_t)batch * SEQL + q0) * DMODEL + head * DKH;
  #pragma unroll
  for (int t = 0; t < 4; ++t)
    #pragma unroll
    for (int v = 0; v < 8; ++v)
      cb[(size_t)(lh * 8 + v) * DMODEL + t * 16 + ln] = (__bf16)(o[t][v] * inv[v]);
}

// ------------------------------------------------------------------
extern "C" void kernel_launch(void* const* d_in, const int* in_sizes, int n_in,
                              void* d_out, int out_size, void* d_ws, size_t ws_size,
                              hipStream_t stream) {
  const float* q    = (const float*)d_in[0];
  const float* k    = (const float*)d_in[1];
  const float* v    = (const float*)d_in[2];
  const int*   mask = (const int*)  d_in[3];
  const float* Wq   = (const float*)d_in[4];
  const float* bq   = (const float*)d_in[5];
  const float* Wk   = (const float*)d_in[6];
  const float* bk   = (const float*)d_in[7];
  const float* Wv   = (const float*)d_in[8];
  const float* bv   = (const float*)d_in[9];
  const float* Wo   = (const float*)d_in[10];
  const float* bo   = (const float*)d_in[11];

  const int    NM  = MTOT * DMODEL;            // 8388608 activation elements
  const size_t NMb = (size_t)NM * 2;           // 16 MB as bf16
  const size_t WMb = (size_t)DMODEL * DMODEL * 2; // 2 MB per weight as bf16

  char* ws = (char*)d_ws;
  __bf16* qb   = (__bf16*)(ws);
  __bf16* kb   = (__bf16*)(ws + NMb);
  __bf16* vb   = (__bf16*)(ws + 2 * NMb);
  __bf16* Wqt  = (__bf16*)(ws + 3 * NMb);
  __bf16* Wkt  = (__bf16*)(ws + 3 * NMb + WMb);
  __bf16* Wvt  = (__bf16*)(ws + 3 * NMb + 2 * WMb);
  __bf16* Wot  = (__bf16*)(ws + 3 * NMb + 3 * WMb);
  __bf16* Qh   = (__bf16*)(ws + 3 * NMb + 4 * WMb);
  __bf16* Kh   = (__bf16*)(ws + 4 * NMb + 4 * WMb);
  __bf16* Vtr  = (__bf16*)(ws + 5 * NMb + 4 * WMb);
  __bf16* ctx  = (__bf16*)(ws + 6 * NMb + 4 * WMb);
  // total workspace use: 7*16MB + 8MB = 120 MB

  // 1) convert activations + weights to bf16 (weights transposed)
  cvt_bf16_kernel<<<(NM + 255) / 256, 256, 0, stream>>>(q, qb, NM);
  cvt_bf16_kernel<<<(NM + 255) / 256, 256, 0, stream>>>(k, kb, NM);
  cvt_bf16_kernel<<<(NM + 255) / 256, 256, 0, stream>>>(v, vb, NM);
  cvt_wt_kernel<<<(DMODEL * DMODEL) / 256, 256, 0, stream>>>(Wq, Wqt);
  cvt_wt_kernel<<<(DMODEL * DMODEL) / 256, 256, 0, stream>>>(Wk, Wkt);
  cvt_wt_kernel<<<(DMODEL * DMODEL) / 256, 256, 0, stream>>>(Wv, Wvt);
  cvt_wt_kernel<<<(DMODEL * DMODEL) / 256, 256, 0, stream>>>(Wo, Wot);

  // 2) projections (WMMA GEMMs, async-LDS double-buffered)
  dim3 ggrid(MTOT / 128, DMODEL / 128);
  gemm_bf16_kernel<0><<<ggrid, 256, 0, stream>>>(qb, Wqt, bq, (void*)Qh);
  gemm_bf16_kernel<0><<<ggrid, 256, 0, stream>>>(kb, Wkt, bk, (void*)Kh);
  gemm_bf16_kernel<1><<<ggrid, 256, 0, stream>>>(vb, Wvt, bv, (void*)Vtr);

  // 3) flash attention (scores never touch HBM)
  attention_kernel<<<dim3(SEQL / 64, NHEAD, BSZ), 128, 0, stream>>>(Qh, Kh, Vtr, mask, ctx);

  // 4) output projection -> fp32 d_out
  gemm_bf16_kernel<2><<<ggrid, 256, 0, stream>>>(ctx, Wot, bo, d_out);
}
